// BlockSparseAttention_67216238182781
// MI455X (gfx1250) — matrix operand order repaired and checked by reading, not compiled, not run on hardware
//
#include <hip/hip_runtime.h>

// ---------------------------------------------------------------------------
// Problem constants (from reference)
// ---------------------------------------------------------------------------
#define B_   2
#define T_   4096
#define D_   1024
#define H_   16
#define DH_  64
#define W_   512
#define BS_  64
#define KC_  4                 // conv taps
#define NB_  (T_ / BS_)        // 64 query blocks
#define CTX_ (W_ + BS_)        // 576 context keys per block
#define TP_  (W_ + T_)         // padded time extent 4608
#define SCALE_ 0.125f          // 1/sqrt(64)

typedef __bf16 bf16;
typedef __attribute__((ext_vector_type(16))) __bf16       v16bf;
typedef __attribute__((ext_vector_type(8)))  float        v8f;
typedef __attribute__((ext_vector_type(4)))  unsigned int u32x4;

union FragU { v16bf v; u32x4 q[2]; };
static_assert(sizeof(v16bf) == 32, "v16bf must be 32 bytes");

// Low 32 bits of a flat pointer to LDS == byte offset within the workgroup's
// LDS allocation (generic LDS pointers are {aperture_hi32, offset32}).
__device__ __forceinline__ unsigned lds_off(const void* p) {
  return (unsigned)(unsigned long long)p;
}

// ---------------------------------------------------------------------------
// fp32 -> bf16 conversion
// ---------------------------------------------------------------------------
__global__ void f32_to_bf16_kernel(const float* __restrict__ in,
                                   bf16* __restrict__ out, int n) {
  int i = blockIdx.x * blockDim.x + threadIdx.x;
  if (i < n) out[i] = (bf16)in[i];
}

// ---------------------------------------------------------------------------
// Zero the W-wide left padding of kh and vT
// ---------------------------------------------------------------------------
__global__ void zero_pad_kernel(bf16* __restrict__ kh, bf16* __restrict__ vT) {
  int i = blockIdx.x * blockDim.x + threadIdx.x;
  const int n = B_ * H_ * W_ * DH_;
  if (i >= n) return;
  int dh   = i % DH_;
  int rest = i / DH_;
  int t    = rest % W_;
  int bh   = rest / W_;
  kh[((size_t)bh * TP_ + t) * DH_ + dh]  = (bf16)0.0f;
  vT[((size_t)bh * DH_ + dh) * TP_ + t]  = (bf16)0.0f;
}

// ---------------------------------------------------------------------------
// Depthwise causal conv (K=4) over qkv, repack into head layouts:
//   qh[bh, t, dh]   kh[bh, W+t, dh]   vT[bh, dh, W+t]   (all bf16)
// ---------------------------------------------------------------------------
__global__ void dwconv_heads_kernel(const bf16* __restrict__ qkv,
                                    const float* __restrict__ wq,
                                    const float* __restrict__ wk,
                                    const float* __restrict__ wv,
                                    bf16* __restrict__ qh,
                                    bf16* __restrict__ kh,
                                    bf16* __restrict__ vT) {
  int i = blockIdx.x * blockDim.x + threadIdx.x;  // over B*T*D
  if (i >= B_ * T_ * D_) return;
  int e    = i % D_;
  int rest = i / D_;
  int t    = rest % T_;
  int b    = rest / T_;
  float aq = 0.f, ak = 0.f, av = 0.f;
#pragma unroll
  for (int j = 0; j < KC_; ++j) {
    int tt = t - (KC_ - 1) + j;
    if (tt >= 0) {
      size_t row = ((size_t)b * T_ + tt) * (3 * D_);
      float xq = (float)qkv[row + e];
      float xk = (float)qkv[row + D_ + e];
      float xv = (float)qkv[row + 2 * D_ + e];
      aq += xq * wq[e * KC_ + j];
      ak += xk * wk[e * KC_ + j];
      av += xv * wv[e * KC_ + j];
    }
  }
  int h = e / DH_, dh = e % DH_;
  size_t bh = (size_t)b * H_ + h;
  qh[(bh * T_ + t) * DH_ + dh]          = (bf16)aq;
  kh[(bh * TP_ + (W_ + t)) * DH_ + dh]  = (bf16)ak;
  vT[(bh * DH_ + dh) * TP_ + (W_ + t)]  = (bf16)av;
}

// ---------------------------------------------------------------------------
// bf16 GEMM: C[M,N] = A[M,K] * Wt[N,K]^T, fp32 accumulate via WMMA.
// 256 threads = 8 waves (4 M x 2 N), block tile 128x64, K step 32.
// Tiles are staged into LDS with GLOBAL_LOAD_ASYNC_TO_LDS_B128 (ASYNCcnt),
// double-buffered so the copy of step k+1 overlaps the WMMAs of step k.
// Per K-step the block reads 12 KB from global exactly once (vs 32 KB with
// per-wave direct loads).
// ---------------------------------------------------------------------------
template <bool OUT_F32>
__global__ __launch_bounds__(256)
void gemm_bf16_kernel(const bf16* __restrict__ A, const bf16* __restrict__ Wt,
                      float* __restrict__ Cf, bf16* __restrict__ Cb,
                      int M, int N, int Kd) {
  __shared__ bf16 Atile[2][128 * 32];  // 8 KB per buffer, row stride 64 B
  __shared__ bf16 Btile[2][64 * 32];   // 4 KB per buffer, row stride 64 B

  const int tid   = threadIdx.x;
  const int lane  = tid & 31;
  const int wid   = tid >> 5;
  const int waveM = wid & 3, waveN = wid >> 2;
  const int mblk  = blockIdx.x * 128;
  const int nblk  = blockIdx.y * 64;
  const int half  = lane >> 4;
  const int lrow  = lane & 15;

  // Per-thread async-copy assignments (fixed across K-steps):
  // A tile: thread t copies 32 B of row (t>>1) at byte (t&1)*32  (2 x B128)
  // B tile: thread t copies 16 B of row (t>>2) at byte (t&3)*16  (1 x B128)
  const int arow = tid >> 1, abo = (tid & 1) * 32;
  const int brow = tid >> 2, bbo = (tid & 3) * 16;

  auto prefetch = [&](int k, int buf) {
    unsigned la = lds_off(&Atile[buf][0]) + arow * 64 + abo;
    unsigned long long ga =
        (unsigned long long)(A + (size_t)(mblk + arow) * Kd + k) + abo;
    asm volatile(
        "global_load_async_to_lds_b128 %0, %1, off\n\t"
        "global_load_async_to_lds_b128 %0, %1, off offset:16"
        :: "v"(la), "v"(ga) : "memory");
    unsigned lb = lds_off(&Btile[buf][0]) + brow * 64 + bbo;
    unsigned long long gb =
        (unsigned long long)(Wt + (size_t)(nblk + brow) * Kd + k) + bbo;
    asm volatile(
        "global_load_async_to_lds_b128 %0, %1, off"
        :: "v"(lb), "v"(gb) : "memory");
  };

  v8f acc[2][2] = {};
  prefetch(0, 0);

  int buf = 0;
  for (int k = 0; k < Kd; k += 32, buf ^= 1) {
    // Own async copies of buf complete; barrier makes all waves' copies
    // visible and proves everyone is done computing the other buffer.
    asm volatile("s_wait_asynccnt 0x0" ::: "memory");
    __syncthreads();
    if (k + 32 < Kd) prefetch(k + 32, buf ^ 1);

    FragU a[2], bw[2];
#pragma unroll
    for (int ms = 0; ms < 2; ++ms) {
      // A-fragment: lane row = lane%16; elems 0..15 = K[16h .. 16h+15]
      const bf16* ap = &Atile[buf][(waveM * 32 + ms * 16 + lrow) * 32];
      a[ms].q[0] = *(const u32x4*)(ap + 16 * half);
      a[ms].q[1] = *(const u32x4*)(ap + 16 * half + 8);
    }
#pragma unroll
    for (int ns = 0; ns < 2; ++ns) {
      // B-fragment: lane col = lane%16; elems 0..15 = K[16h .. 16h+15]
      const bf16* bp = &Btile[buf][(waveN * 32 + ns * 16 + lrow) * 32 + 16 * half];
      bw[ns].q[0] = *(const u32x4*)(bp);
      bw[ns].q[1] = *(const u32x4*)(bp + 8);
    }
#pragma unroll
    for (int ms = 0; ms < 2; ++ms)
#pragma unroll
      for (int ns = 0; ns < 2; ++ns)
        acc[ms][ns] = __builtin_amdgcn_wmma_f32_16x16x32_bf16(
            false, a[ms].v, false, bw[ns].v, (short)0, acc[ms][ns], false, false);
  }

  // C layout: VGPR r -> row r (lanes 0..15) / row r+8 (lanes 16..31), col = lane%16
#pragma unroll
  for (int ms = 0; ms < 2; ++ms)
#pragma unroll
    for (int ns = 0; ns < 2; ++ns)
#pragma unroll
      for (int r = 0; r < 8; ++r) {
        int row = mblk + waveM * 32 + ms * 16 + r + 8 * half;
        int col = nblk + waveN * 32 + ns * 16 + lrow;
        if constexpr (OUT_F32)
          Cf[(size_t)row * N + col] = acc[ms][ns][r];
        else
          Cb[(size_t)row * N + col] = (bf16)acc[ms][ns][r];
      }
}

// ---------------------------------------------------------------------------
// Sliding-window block attention (flash-style online softmax).
// One block per (b, h, query-block n); 4 waves, each owning 16 query rows.
// ---------------------------------------------------------------------------
__global__ __launch_bounds__(128)
void attn_kernel(const bf16* __restrict__ qh, const bf16* __restrict__ kh,
                 const bf16* __restrict__ vT, bf16* __restrict__ attn) {
  __shared__ bf16 pstage[4][16 * 32];  // per-wave 16x32 bf16 P tile

  const int lane = threadIdx.x & 31;
  const int wid  = threadIdx.x >> 5;
  const int half = lane >> 4;
  const int lrow = lane & 15;
  const int blk  = blockIdx.x;
  const int n    = blk % NB_;
  const int bh   = blk / NB_;
  const int b    = bh / H_, h = bh % H_;

  // Q fragments for the two dh reduction steps (0..31, 32..63)
  FragU aq[2];
  {
    const bf16* qp = qh + ((size_t)bh * T_ + n * BS_ + wid * 16 + lrow) * DH_;
#pragma unroll
    for (int s = 0; s < 2; ++s) {
      aq[s].q[0] = *(const u32x4*)(qp + 32 * s + 16 * half);
      aq[s].q[1] = *(const u32x4*)(qp + 32 * s + 16 * half + 8);
    }
  }

  v8f accO[4] = {};
  float mrow[8], lsum[8];
#pragma unroll
  for (int r = 0; r < 8; ++r) { mrow[r] = -3.0e38f; lsum[r] = 0.f; }

  const int lo = (n < 8) ? (W_ - n * BS_) : 0;  // smallest valid context column

  for (int kc = 0; kc < CTX_; kc += 32) {
    if (kc + 32 <= lo) continue;  // uniform across block: no barrier divergence

    // ---- scores: S = Q * K^T for two 16-key subtiles
    v8f s0 = {}, s1 = {};
#pragma unroll
    for (int s = 0; s < 2; ++s) {
      FragU bk0, bk1;
      const bf16* kp0 =
          kh + ((size_t)bh * TP_ + n * BS_ + kc + lrow) * DH_ + 32 * s + 16 * half;
      const bf16* kp1 =
          kh + ((size_t)bh * TP_ + n * BS_ + kc + 16 + lrow) * DH_ + 32 * s + 16 * half;
      bk0.q[0] = *(const u32x4*)kp0;
      bk0.q[1] = *(const u32x4*)(kp0 + 8);
      bk1.q[0] = *(const u32x4*)kp1;
      bk1.q[1] = *(const u32x4*)(kp1 + 8);
      s0 = __builtin_amdgcn_wmma_f32_16x16x32_bf16(false, aq[s].v, false, bk0.v,
                                                   (short)0, s0, false, false);
      s1 = __builtin_amdgcn_wmma_f32_16x16x32_bf16(false, aq[s].v, false, bk1.v,
                                                   (short)0, s1, false, false);
    }

    // ---- mask + scale + online softmax update (rows live in (vgpr r, lane-half))
    float p0[8], p1[8];
#pragma unroll
    for (int r = 0; r < 8; ++r) {
      int qb = wid * 16 + r + 8 * half;  // query index within block [0,64)
      int c0 = kc + lrow;                // context column of s0
      int c1 = c0 + 16;
      float v0 = ((c0 >= lo) && (c0 <= qb + W_)) ? s0[r] * SCALE_ : -1.0e30f;
      float v1 = ((c1 >= lo) && (c1 <= qb + W_)) ? s1[r] * SCALE_ : -1.0e30f;
      float mt = fmaxf(v0, v1);
#pragma unroll
      for (int off = 8; off >= 1; off >>= 1)
        mt = fmaxf(mt, __shfl_xor(mt, off, 32));  // stays within 16-lane half
      float mnew = fmaxf(mrow[r], mt);
      float sf   = __expf(mrow[r] - mnew);
      float e0   = __expf(v0 - mnew);
      float e1   = __expf(v1 - mnew);
      float ps   = e0 + e1;
#pragma unroll
      for (int off = 8; off >= 1; off >>= 1) ps += __shfl_xor(ps, off, 32);
      lsum[r] = lsum[r] * sf + ps;
      mrow[r] = mnew;
#pragma unroll
      for (int j = 0; j < 4; ++j) accO[j][r] *= sf;
      p0[r] = e0;
      p1[r] = e1;
    }

    // ---- stage P (C layout) to LDS, reload as 16x32 A fragment
    __syncthreads();
#pragma unroll
    for (int r = 0; r < 8; ++r) {
      int row = r + 8 * half;
      pstage[wid][row * 32 + lrow]      = (bf16)p0[r];
      pstage[wid][row * 32 + 16 + lrow] = (bf16)p1[r];
    }
    __syncthreads();
    FragU pa;
    {
      const bf16* pp = &pstage[wid][lrow * 32];
      pa.q[0] = *(const u32x4*)(pp + 16 * half);
      pa.q[1] = *(const u32x4*)(pp + 16 * half + 8);
    }

    // ---- O += P x V (V transposed: per-lane contiguous 32B B-fragment loads)
#pragma unroll
    for (int j = 0; j < 4; ++j) {
      FragU bv;
      const bf16* vp =
          vT + ((size_t)bh * DH_ + j * 16 + lrow) * TP_ + n * BS_ + kc + 16 * half;
      bv.q[0] = *(const u32x4*)vp;
      bv.q[1] = *(const u32x4*)(vp + 8);
      accO[j] = __builtin_amdgcn_wmma_f32_16x16x32_bf16(false, pa.v, false, bv.v,
                                                        (short)0, accO[j], false, false);
    }
  }

  // ---- normalize and store to [B, T, D] bf16 (input to output projection)
  const int tbase = n * BS_ + wid * 16;
#pragma unroll
  for (int r = 0; r < 8; ++r) {
    float inv  = 1.0f / lsum[r];
    int tglob  = tbase + r + 8 * half;
    size_t row = ((size_t)b * T_ + tglob) * D_ + h * DH_;
#pragma unroll
    for (int j = 0; j < 4; ++j)
      attn[row + j * 16 + lrow] = (bf16)(accO[j][r] * inv);
  }
}

// ---------------------------------------------------------------------------
// Host-side orchestration
// ---------------------------------------------------------------------------
extern "C" void kernel_launch(void* const* d_in, const int* in_sizes, int n_in,
                              void* d_out, int out_size, void* d_ws, size_t ws_size,
                              hipStream_t stream) {
  const float* x     = (const float*)d_in[0];
  const float* qkv_w = (const float*)d_in[1];
  const float* wq    = (const float*)d_in[2];
  const float* wk    = (const float*)d_in[3];
  const float* wv    = (const float*)d_in[4];
  const float* out_w = (const float*)d_in[5];
  float* out = (float*)d_out;

  char* ws = (char*)d_ws;
  size_t off = 0;
  auto carve = [&](size_t bytes) -> void* {
    void* p = ws + off;
    off += (bytes + 255) & ~(size_t)255;
    return p;
  };
  bf16* xb    = (bf16*)carve((size_t)B_ * T_ * D_ * 2);        // 16.8 MB
  bf16* qkvwb = (bf16*)carve((size_t)3 * D_ * D_ * 2);         //  6.3 MB
  bf16* outwb = (bf16*)carve((size_t)D_ * D_ * 2);             //  2.1 MB
  bf16* qkvb  = (bf16*)carve((size_t)B_ * T_ * 3 * D_ * 2);    // 50.3 MB
  bf16* qh    = (bf16*)carve((size_t)B_ * H_ * T_ * DH_ * 2);  // 16.8 MB
  bf16* kh    = (bf16*)carve((size_t)B_ * H_ * TP_ * DH_ * 2); // 18.9 MB
  bf16* vT    = (bf16*)carve((size_t)B_ * H_ * DH_ * TP_ * 2); // 18.9 MB
  bf16* attnb = qkvb;  // qkv buffer is dead after conv; reuse for attention out

  const int n1 = B_ * T_ * D_;
  f32_to_bf16_kernel<<<(n1 + 255) / 256, 256, 0, stream>>>(x, xb, n1);
  const int n2 = 3 * D_ * D_;
  f32_to_bf16_kernel<<<(n2 + 255) / 256, 256, 0, stream>>>(qkv_w, qkvwb, n2);
  const int n3 = D_ * D_;
  f32_to_bf16_kernel<<<(n3 + 255) / 256, 256, 0, stream>>>(out_w, outwb, n3);

  // QKV projection: [8192,1024] x [1024,3072]
  dim3 g1(B_ * T_ / 128, 3 * D_ / 64);
  gemm_bf16_kernel<false><<<g1, 256, 0, stream>>>(
      xb, qkvwb, nullptr, qkvb, B_ * T_, 3 * D_, D_);

  const int npad = B_ * H_ * W_ * DH_;
  zero_pad_kernel<<<(npad + 255) / 256, 256, 0, stream>>>(kh, vT);
  dwconv_heads_kernel<<<(n1 + 255) / 256, 256, 0, stream>>>(
      qkvb, wq, wk, wv, qh, kh, vT);

  // Attention: one block per (b, h, query-block)
  attn_kernel<<<B_ * H_ * NB_, 128, 0, stream>>>(qh, kh, vT, attnb);

  // Output projection: [8192,1024] x [1024,1024] -> fp32
  dim3 g2(B_ * T_ / 128, D_ / 64);
  gemm_bf16_kernel<true><<<g2, 256, 0, stream>>>(
      attnb, outwb, out, nullptr, B_ * T_, D_, D_);
}
